// MultiheadAttention_28613072126646
// MI455X (gfx1250) — matrix operand order repaired and checked by reading, not compiled, not run on hardware
//
#include <hip/hip_runtime.h>

// ---------------------------------------------------------------------------
// MultiheadAttention (no softmax) for MI455X / gfx1250, wave32 + WMMA bf16.
// L=1024, B=8, E=1024, H=16, D=64.
// d_out = [ out (L*B*E = 8388608 f32) | attn (B*H*L*L = 134217728 f32) ]
// ---------------------------------------------------------------------------

typedef __bf16 bf16_t;
typedef __attribute__((ext_vector_type(16))) __bf16 v16bf;
typedef __attribute__((ext_vector_type(8)))  __bf16 v8bf;
typedef __attribute__((ext_vector_type(4)))  __bf16 v4bf;
typedef __attribute__((ext_vector_type(8)))  float  v8f;
typedef __attribute__((ext_vector_type(4)))  float  v4f;

static __device__ __forceinline__ v8f wmma_bf16(v16bf a, v16bf b, v8f c) {
  // D = A(16x32) * B(32x16) + C, f32 accumulate
  return __builtin_amdgcn_wmma_f32_16x16x32_bf16(false, a, false, b,
                                                 (short)0, c, false, false);
}

static __device__ __forceinline__ v8f vzero8() {
  v8f z = {0.f, 0.f, 0.f, 0.f, 0.f, 0.f, 0.f, 0.f};
  return z;
}

// A-matrix 16x32 bf16 fragment. Lane (m = lane&15, half = lane>>4) holds
// K = k0 + half*8 + [0..8)  in elements 0..7, and
// K = k0 + 16 + half*8 + [0..8) in elements 8..15 (row m of A).
static __device__ __forceinline__ v16bf load_a_frag(const bf16_t* row, int k0, int half) {
  union { v16bf v; struct { v8bf lo, hi; } s; } u;
  u.s.lo = *(const v8bf*)(row + k0 + half * 8);
  u.s.hi = *(const v8bf*)(row + k0 + 16 + half * 8);
  return u.v;
}

// B-matrix 32x16 bf16 fragment. Lane (n = lane&15, half) holds column n,
// K = k0 + half*8 + [0..8) in elements 0..7, K = k0 + 16 + half*8 + [0..8) in 8..15. `row` = pointer to the K-contiguous
// vector that forms column n of B (e.g. row n of W for C = A*W^T).
static __device__ __forceinline__ v16bf load_b_frag(const bf16_t* row, int k0, int half) {
  union { v16bf v; struct { v8bf lo, hi; } s; } u;
  u.s.lo = *(const v8bf*)(row + k0 + half * 8);
  u.s.hi = *(const v8bf*)(row + k0 + 16 + half * 8);
  return u.v;
}

// ---------------------------------------------------------------------------
// f32 -> bf16 bulk convert
// ---------------------------------------------------------------------------
__global__ void cvt_f32_bf16(const float* __restrict__ src,
                             bf16_t* __restrict__ dst, int n4) {
  int i = blockIdx.x * blockDim.x + threadIdx.x;
  if (i < n4) {
    v4f x = ((const v4f*)src)[i];
    v4bf o;
    o[0] = (bf16_t)x[0]; o[1] = (bf16_t)x[1];
    o[2] = (bf16_t)x[2]; o[3] = (bf16_t)x[3];
    ((v4bf*)dst)[i] = o;
  }
}

// ---------------------------------------------------------------------------
// C[m,n] = sum_k A[m,k]*W[n,k] + bias[n]
// A: (8192 x 1024) bf16 row-major, W: (Nrel x 1024) bf16 row-major.
// Workgroup = 128 threads (4 waves), WG tile 64(M) x 128(N), wave 32x64.
// Output either bf16 (pitch ldc) or f32 (pitch ldc).
// ---------------------------------------------------------------------------
template <bool OUT_F32>
__global__ __launch_bounds__(128) void gemm_nt(
    const bf16_t* __restrict__ A, const bf16_t* __restrict__ W,
    const float* __restrict__ bias, void* __restrict__ Cout, int ldc) {
  constexpr int K = 1024;
  __shared__ float stage[4][32][68];  // per-wave 32x64 (+4 pad), 16B-aligned pitch

  const int tid = threadIdx.x;
  const int wave = tid >> 5, lane = tid & 31;
  const int mm = lane & 15, half = lane >> 4;
  const int m_base = blockIdx.y * 64 + (wave >> 1) * 32;
  const int n_base = blockIdx.x * 128 + (wave & 1) * 64;

  const bf16_t* arow0 = A + (size_t)(m_base + mm) * K;
  const bf16_t* arow1 = arow0 + (size_t)16 * K;
  const bf16_t* wrow = W + (size_t)(n_base + mm) * K;

  v8f acc[2][4];
#pragma unroll
  for (int mt = 0; mt < 2; ++mt)
#pragma unroll
    for (int t = 0; t < 4; ++t) acc[mt][t] = vzero8();

  for (int k0 = 0; k0 < K; k0 += 32) {
    v16bf a0 = load_a_frag(arow0, k0, half);
    v16bf a1 = load_a_frag(arow1, k0, half);
#pragma unroll
    for (int t = 0; t < 4; ++t) {
      v16bf bt = load_b_frag(wrow + (size_t)t * 16 * K, k0, half);
      acc[0][t] = wmma_bf16(a0, bt, acc[0][t]);
      acc[1][t] = wmma_bf16(a1, bt, acc[1][t]);
    }
  }

  // bias (column n = n_base + t*16 + mm for every element of the C fragment)
#pragma unroll
  for (int t = 0; t < 4; ++t) {
    float bv = bias[n_base + t * 16 + mm];
#pragma unroll
    for (int r = 0; r < 8; ++r) { acc[0][t][r] += bv; acc[1][t][r] += bv; }
  }

  // stage C fragments (row = mt*16 + half*8 + r, col = t*16 + mm)
#pragma unroll
  for (int mt = 0; mt < 2; ++mt)
#pragma unroll
    for (int t = 0; t < 4; ++t)
#pragma unroll
      for (int r = 0; r < 8; ++r)
        stage[wave][mt * 16 + half * 8 + r][t * 16 + mm] = acc[mt][t][r];

  asm volatile("s_wait_dscnt 0" ::: "memory");  // wave-local LDS RAW

  // coalesced store: 32 rows x 64 cols per wave = 512 float4 chunks
#pragma unroll
  for (int i = 0; i < 16; ++i) {
    int chunk = i * 32 + lane;
    int row = chunk >> 4;
    int cc = (chunk & 15) * 4;
    v4f d = *(const v4f*)&stage[wave][row][cc];
    size_t gidx = (size_t)(m_base + row) * (size_t)ldc + (size_t)(n_base + cc);
    if constexpr (OUT_F32) {
      *(v4f*)((float*)Cout + gidx) = d;
    } else {
      v4bf o;
      o[0] = (bf16_t)d[0]; o[1] = (bf16_t)d[1];
      o[2] = (bf16_t)d[2]; o[3] = (bf16_t)d[3];
      *(v4bf*)((bf16_t*)Cout + gidx) = o;
    }
  }
}

// ---------------------------------------------------------------------------
// Fused attention (no softmax):
//   S = q k^T * 0.125  -> written to attn (f32, once, coalesced)
//   O = S v            -> accumulated in registers, stored bf16 to out1
// qkv: (8192 x 3072) bf16, cols [0,1024)=q, [1024,2048)=k, [2048,3072)=v,
// row index = l*8 + b, head h occupies 64 cols.
// Grid: 2048 blocks = 128 (b,h) x 16 m-blocks; block = 128 thr (4 waves),
// each wave owns 16 query rows; loop over key blocks of 64.
// ---------------------------------------------------------------------------
__global__ __launch_bounds__(128) void attn_fused(
    const bf16_t* __restrict__ qkv, float* __restrict__ attn,
    bf16_t* __restrict__ out1) {
  constexpr int LDQ = 3072;
  __shared__ float sstage[4][16][68];   // per-wave 16x64 f32 scores (+pad)
  __shared__ bf16_t vtile[64][88];      // v^T block: [d][s_local] (+pad)

  const int tid = threadIdx.x, wave = tid >> 5, lane = tid & 31;
  const int mm = lane & 15, half = lane >> 4;
  const int bh = blockIdx.x >> 4, mb = blockIdx.x & 15;
  const int b = bh >> 4, h = bh & 15;
  const int m0 = mb * 64 + wave * 16;  // wave's first query row

  // q A-fragments for this wave's 16 rows (K = d = 64 -> two 16x32 chunks)
  const bf16_t* qrow = qkv + (size_t)((m0 + mm) * 8 + b) * LDQ + h * 64;
  v16bf aq0 = load_a_frag(qrow, 0, half);
  v16bf aq1 = load_a_frag(qrow, 32, half);

  v8f oacc[4] = {vzero8(), vzero8(), vzero8(), vzero8()};

  const bf16_t* kbase = qkv + 1024 + h * 64;
  const bf16_t* vbase = qkv + 2048 + h * 64;
  float* arow_base = attn + ((size_t)bh * 1024 + (size_t)m0) * 1024;

  for (int sb = 0; sb < 1024; sb += 64) {
    __syncthreads();  // protect vtile WAR vs previous iteration's reads
    // cooperatively load v block (64 s x 64 d) transposed: vtile[d][s_local]
#pragma unroll
    for (int i = 0; i < 4; ++i) {
      int chunk = i * 128 + tid;          // 512 chunks of 8 bf16
      int s_local = chunk >> 3;
      int dch = (chunk & 7) * 8;
      v8bf vv = *(const v8bf*)(vbase + (size_t)((sb + s_local) * 8 + b) * LDQ + dch);
#pragma unroll
      for (int j = 0; j < 8; ++j) vtile[dch + j][s_local] = vv[j];
    }
    __syncthreads();

    // S tiles: 4 n-tiles of 16 keys, K-loop over d (2 x 16x16x32 WMMA)
#pragma unroll
    for (int t = 0; t < 4; ++t) {
      const bf16_t* krow = kbase + (size_t)((sb + t * 16 + mm) * 8 + b) * LDQ;
      v8f s = vzero8();
      s = wmma_bf16(aq0, load_b_frag(krow, 0, half), s);
      s = wmma_bf16(aq1, load_b_frag(krow, 32, half), s);
#pragma unroll
      for (int r = 0; r < 8; ++r)
        sstage[wave][half * 8 + r][t * 16 + mm] = s[r] * 0.125f;
    }
    asm volatile("s_wait_dscnt 0" ::: "memory");

    // coalesced attn store: 16 rows x 64 cols = 256 float4 chunks per wave
#pragma unroll
    for (int i = 0; i < 8; ++i) {
      int chunk = i * 32 + lane;
      int row = chunk >> 4;
      int cc = (chunk & 15) * 4;
      v4f d = *(const v4f*)&sstage[wave][row][cc];
      *(v4f*)(arow_base + (size_t)row * 1024 + sb + cc) = d;
    }

    // O += S * v : A = staged scores (cvt f32->bf16), B = vtile columns
    v16bf pa[2];
#pragma unroll
    for (int c = 0; c < 2; ++c) {
      const float* sp = &sstage[wave][mm][c * 32];
      v4f f0 = *(const v4f*)(sp + half * 8);
      v4f f1 = *(const v4f*)(sp + half * 8 + 4);
      v4f f2 = *(const v4f*)(sp + 16 + half * 8);
      v4f f3 = *(const v4f*)(sp + 16 + half * 8 + 4);
      union { v16bf v; bf16_t e[16]; } u;
#pragma unroll
      for (int j = 0; j < 4; ++j) {
        u.e[j]      = (bf16_t)f0[j];
        u.e[4 + j]  = (bf16_t)f1[j];
        u.e[8 + j]  = (bf16_t)f2[j];
        u.e[12 + j] = (bf16_t)f3[j];
      }
      pa[c] = u.v;
    }
#pragma unroll
    for (int dt = 0; dt < 4; ++dt) {
#pragma unroll
      for (int c = 0; c < 2; ++c) {
        union { v16bf v; struct { v8bf lo, hi; } s; } ub;
        const bf16_t* vp = &vtile[dt * 16 + mm][c * 32 + half * 8];
        ub.s.lo = *(const v8bf*)vp;
        ub.s.hi = *(const v8bf*)(vp + 16);
        oacc[dt] = wmma_bf16(pa[c], ub.v, oacc[dt]);
      }
    }
  }

  // store O (16 x 64) as bf16 into out1 (L,B,E) via staging
#pragma unroll
  for (int dt = 0; dt < 4; ++dt)
#pragma unroll
    for (int r = 0; r < 8; ++r)
      sstage[wave][half * 8 + r][dt * 16 + mm] = oacc[dt][r];
  asm volatile("s_wait_dscnt 0" ::: "memory");
#pragma unroll
  for (int i = 0; i < 8; ++i) {
    int chunk = i * 32 + lane;
    int row = chunk >> 4;
    int cc = (chunk & 15) * 4;
    v4f d = *(const v4f*)&sstage[wave][row][cc];
    v4bf o;
    o[0] = (bf16_t)d[0]; o[1] = (bf16_t)d[1];
    o[2] = (bf16_t)d[2]; o[3] = (bf16_t)d[3];
    *(v4bf*)(out1 + (size_t)((m0 + row) * 8 + b) * 1024 + h * 64 + cc) = o;
  }
}

// ---------------------------------------------------------------------------
extern "C" void kernel_launch(void* const* d_in, const int* in_sizes, int n_in,
                              void* d_out, int out_size, void* d_ws,
                              size_t ws_size, hipStream_t stream) {
  (void)in_sizes; (void)n_in; (void)out_size; (void)ws_size;

  const float* q_f  = (const float*)d_in[0];
  const float* k_f  = (const float*)d_in[1];
  const float* v_f  = (const float*)d_in[2];
  const float* w_in = (const float*)d_in[3];
  const float* b_in = (const float*)d_in[4];
  const float* w_out = (const float*)d_in[5];
  const float* b_out = (const float*)d_in[6];

  constexpr size_t MB = 1u << 20;
  constexpr size_t NXY = 1024u * 8u * 1024u;  // 8388608 elems of x / out
  constexpr size_t EE = 1024u * 1024u;

  char* ws = (char*)d_ws;
  bf16_t* xq    = (bf16_t*)(ws + 0 * MB);    // 16 MB
  bf16_t* xk    = (bf16_t*)(ws + 16 * MB);   // 16 MB
  bf16_t* xv    = (bf16_t*)(ws + 32 * MB);   // 16 MB
  bf16_t* winb  = (bf16_t*)(ws + 48 * MB);   // 6 MB (3072x1024)
  bf16_t* woutb = (bf16_t*)(ws + 54 * MB);   // 2 MB (1024x1024)
  bf16_t* qkvb  = (bf16_t*)(ws + 56 * MB);   // 48 MB (8192x3072)
  bf16_t* out1  = xq;                        // alias: xq dead after q-proj

  // 1) convert inputs/weights to bf16
  cvt_f32_bf16<<<(int)(NXY / 4 / 256), 256, 0, stream>>>(q_f, xq, (int)(NXY / 4));
  cvt_f32_bf16<<<(int)(NXY / 4 / 256), 256, 0, stream>>>(k_f, xk, (int)(NXY / 4));
  cvt_f32_bf16<<<(int)(NXY / 4 / 256), 256, 0, stream>>>(v_f, xv, (int)(NXY / 4));
  cvt_f32_bf16<<<(int)(3 * EE / 4 / 256), 256, 0, stream>>>(w_in, winb, (int)(3 * EE / 4));
  cvt_f32_bf16<<<(int)(EE / 4 / 256), 256, 0, stream>>>(w_out, woutb, (int)(EE / 4));

  // 2) in-projections: q,k,v into packed qkv (pitch 3072)
  dim3 gblock(128);
  dim3 ggrid(8, 128);  // N=1024 -> 8 tiles of 128, M=8192 -> 128 tiles of 64
  gemm_nt<false><<<ggrid, gblock, 0, stream>>>(xq, winb,          b_in,        qkvb,        3072);
  gemm_nt<false><<<ggrid, gblock, 0, stream>>>(xk, winb + EE,     b_in + 1024, qkvb + 1024, 3072);
  gemm_nt<false><<<ggrid, gblock, 0, stream>>>(xv, winb + 2 * EE, b_in + 2048, qkvb + 2048, 3072);

  // 3) fused scores + attn*v ; attn output lives after `out` in d_out
  float* attn_out = (float*)d_out + NXY;
  attn_fused<<<2048, 128, 0, stream>>>(qkvb, attn_out, out1);

  // 4) out-projection (f32 + bias) into d_out[0 : NXY)
  dim3 ogrid(8, 128);
  gemm_nt<true><<<ogrid, gblock, 0, stream>>>(out1, woutb, b_out, d_out, 1024);
}
